// DeepFM_43224550867489
// MI455X (gfx1250) — compile-verified, matrix operand order checked by reading, not run on hardware
//
#include <hip/hip_runtime.h>
#include <hip/hip_bf16.h>

typedef __attribute__((ext_vector_type(16))) __bf16 v16bf;
typedef __attribute__((ext_vector_type(8)))  __bf16 v8bf;
typedef __attribute__((ext_vector_type(8)))  float  v8f;

#define TSZ   100          // tokens per sample
#define FD    64           // embedding dim
#define FH    32           // deep hidden dim
#define TP    112          // padded token rows (7 tiles of 16)
#define NT    7            // row tiles
#define STR16 72           // bf16 LDS row stride (16B aligned, destaggered banks)
#define STRF  68           // f32  LDS row stride (16B aligned)
#define DEEP_LEN  (TSZ*FH)            // 3200  (concat: [deep, fm, wide])
#define FM_BASE   DEEP_LEN            // 3200
#define FM_LEN    (TSZ*(TSZ-1)/2)     // 4950
#define WIDE_BASE (FM_BASE+FM_LEN)    // 8150

__device__ __forceinline__ unsigned short f32_to_bf16(float f) {
    unsigned int u = __float_as_uint(f);
    u += 0x7fffu + ((u >> 16) & 1u);          // round-to-nearest-even
    return (unsigned short)(u >> 16);
}

// Build a v16bf WMMA fragment for one 16x32 K-window from a row-major bf16 row in LDS.
// CDNA5 16-bit A layout: lanes 0-15 hold K {0..7,16..23} of the window, lanes 16-31
// hold K {8..15,24..31}. B (column-per-lane) mirrors it, so one loader serves both.
__device__ __forceinline__ v16bf load_frag(const unsigned short* row, int kstep, int lane) {
    const int kbase = kstep * 32 + ((lane & 16) ? 8 : 0);
    const v8bf lo = *reinterpret_cast<const v8bf*>(row + kbase);
    const v8bf hi = *reinterpret_cast<const v8bf*>(row + kbase + 16);
    v16bf r;
#pragma unroll
    for (int i = 0; i < 8; ++i) { r[i] = lo[i]; r[i + 8] = hi[i]; }
    return r;
}

__global__ __launch_bounds__(256)
void deepfm_fused(const int* __restrict__ x, const float* __restrict__ emb,
                  const float* __restrict__ w_deep, const float* __restrict__ b_deep,
                  const float* __restrict__ w_ffn, const float* __restrict__ b_ffn,
                  float* __restrict__ out)
{
    __shared__ int            sIdx[TSZ];
    __shared__ float          sEf[TSZ * STRF];        // f32 copy of E (async-DMA'd, exact for wide)
    __shared__ unsigned short sE16[TP * STR16];       // bf16 copy of E (zero-padded rows)
    __shared__ unsigned short sWdT[FH * STR16];       // w_deep transposed, bf16: [col][k]
    __shared__ float          sSum;

    const int b    = blockIdx.x;
    const int tid  = threadIdx.x;
    const int lane = tid & 31;
    const int wave = tid >> 5;

    if (tid == 0) sSum = 0.f;
    if (tid < TSZ) sIdx[tid] = x[b * TSZ + tid];

    // w_deep [64,32] row-major -> sWdT[c*STR16 + k] = w_deep[k,c] (bf16)
    for (int u = tid; u < FH * FD; u += 256) {
        const int c = u >> 6, k = u & 63;
        sWdT[c * STR16 + k] = f32_to_bf16(w_deep[k * FH + c]);
    }
    // zero the padding rows 100..111 (keeps WMMA tails clean)
    for (int u = tid; u < (TP - TSZ) * STR16; u += 256)
        sE16[TSZ * STR16 + u] = 0;
    __syncthreads();

    // ---- Gather embedding rows straight into LDS via async DMA (ASYNCcnt path).
    // emb table is 25.6MB -> fully L2-resident; per-lane 16B granules.
    for (int u = tid; u < TSZ * (FD / 4); u += 256) {
        const int t = u >> 4;
        const int c = (u & 15) * 4;                                   // float offset in row
        const unsigned lds = (unsigned)(unsigned long long)&sEf[t * STRF + c];
        const unsigned long long ga =
            (unsigned long long)(emb + (unsigned long long)sIdx[t] * FD + c);
        asm volatile("global_load_async_to_lds_b128 %0, %1, off"
                     :: "v"(lds), "v"(ga) : "memory");
    }
    asm volatile("s_wait_asynccnt 0" ::: "memory");
    __syncthreads();

    float acc = 0.f;

    // ---- bf16 conversion pass fused with the wide dot-product (one sweep over E) ----
    for (int u = tid; u < TSZ * FD; u += 256) {
        const int t = u >> 6, f = u & 63;
        const float ev = sEf[t * STRF + f];
        sE16[t * STR16 + f] = f32_to_bf16(ev);
        acc = fmaf(ev, w_ffn[WIDE_BASE + u], acc);
    }
    __syncthreads();

    // ---- tile tasks: 0..27 = FM Gram tiles (ti<=tj), 28..41 = deep tiles ----
    const int rowm = lane & 15;
    const int voff = (lane >> 4) * 8;       // C layout: M = 16*ti + v + voff, N = 16*tj + rowm
    for (int task = wave; task < 42; task += 8) {
        if (task < 28) {
            int u = task, ti = 0;
            while (u >= NT - ti) { u -= NT - ti; ++ti; }
            const int tj = ti + u;
            const unsigned short* arow = &sE16[(ti * 16 + rowm) * STR16];
            const unsigned short* brow = &sE16[(tj * 16 + rowm) * STR16];
            v8f g = {};
#pragma unroll
            for (int ks = 0; ks < 2; ++ks) {
                const v16bf a  = load_frag(arow, ks, lane);
                const v16bf bb = load_frag(brow, ks, lane);
                g = __builtin_amdgcn_wmma_f32_16x16x32_bf16(false, a, false, bb,
                                                            (short)0, g, false, false);
            }
            const int j = tj * 16 + rowm;
#pragma unroll
            for (int v = 0; v < 8; ++v) {
                const int i = ti * 16 + v + voff;
                const bool valid = (i < j) & (j < TSZ);
                const int idx = valid ? (i * (2 * TSZ - 1 - i)) / 2 + (j - i - 1) : 0;
                const float wv = w_ffn[FM_BASE + idx];          // unconditional, safe addr
                acc = fmaf(g[v], valid ? wv : 0.f, acc);        // mask via select, no branch
            }
        } else {
            const int u  = task - 28;
            const int ti = u >> 1;
            const int n0 = (u & 1) * 16;
            const unsigned short* arow = &sE16[(ti * 16 + rowm) * STR16];
            const unsigned short* brow = &sWdT[(n0 + rowm) * STR16];
            v8f d = {};
#pragma unroll
            for (int ks = 0; ks < 2; ++ks) {
                const v16bf a  = load_frag(arow, ks, lane);
                const v16bf bb = load_frag(brow, ks, lane);
                d = __builtin_amdgcn_wmma_f32_16x16x32_bf16(false, a, false, bb,
                                                            (short)0, d, false, false);
            }
            const int   n  = n0 + rowm;
            const float bd = b_deep[n];
#pragma unroll
            for (int v = 0; v < 8; ++v) {
                const int m  = ti * 16 + v + voff;
                const int mc = (m < TSZ) ? m : (TSZ - 1);
                const float wv = w_ffn[mc * FH + n];            // unconditional, safe addr
                const float h  = fmaxf(d[v] + bd, 0.f);
                acc = fmaf(h, (m < TSZ) ? wv : 0.f, acc);
            }
        }
    }

    atomicAdd(&sSum, acc);                              // ds_add_f32
    __syncthreads();
    if (tid == 0) {
        const float t = sSum + b_ffn[0];
        out[b] = 1.f / (1.f + __expf(-t));
    }
}

extern "C" void kernel_launch(void* const* d_in, const int* in_sizes, int n_in,
                              void* d_out, int out_size, void* d_ws, size_t ws_size,
                              hipStream_t stream) {
    const int*   x      = (const int*)d_in[0];
    const float* emb    = (const float*)d_in[1];
    const float* w_deep = (const float*)d_in[2];
    const float* b_deep = (const float*)d_in[3];
    const float* w_ffn  = (const float*)d_in[4];
    const float* b_ffn  = (const float*)d_in[5];
    const int bs = in_sizes[0] / TSZ;                  // 8192
    deepfm_fused<<<bs, 256, 0, stream>>>(x, emb, w_deep, b_deep, w_ffn, b_ffn, (float*)d_out);
}
